// MyelinSRNet_38036230374133
// MI455X (gfx1250) — compile-verified, matrix operand order checked by reading, not run on hardware
//
#include <hip/hip_runtime.h>
#include <hip/hip_bf16.h>
#include <cstdint>

// ---------------------------------------------------------------------------
// MyelinSRNet forward on gfx1250 (MI455X).
//
// Dominant compute: 8x pointwise 64x64 1x1 convs (expert res-blocks) mapped
// to v_wmma_f32_16x16x32_f16 (f32 accumulate). The depthwise producer writes
// its output as f16 in pixel-major tiles ([B, H*W, 64]) -- halves the hottest
// intermediate stream and makes GEMM tiles contiguous 128B rows, which are
// staged into LDS with gfx1250 async copies (global_load_async_to_lds_b128 +
// s_wait_asynccnt). Other activations stay fp32 (L2-resident on 192MB L2).
// ---------------------------------------------------------------------------

#define BB 4
#define CC 64
#define HH 256
#define WW 256
#define HWHW (HH * WW)          // 65536
#define NEXP 4
#define NBLK 2
#define H2 (HH * 2)
#define W2 (WW * 2)

typedef __attribute__((ext_vector_type(16))) _Float16 v16h;
typedef __attribute__((ext_vector_type(8)))  float    v8f;

__device__ __forceinline__ float sigmoidf_(float x) { return 1.0f / (1.0f + expf(-x)); }

// ---------------------------------------------------------------------------
// TWN ternarization: delta = 0.75*mean|w|; alpha = mean|w| over survivors.
// One block per weight slice.
// ---------------------------------------------------------------------------
__global__ void ternarize_kernel(const float* __restrict__ src,
                                 float* __restrict__ dst, int n) {
    __shared__ float red[256];
    __shared__ float stats[2];
    const int slice = blockIdx.x;
    const float* s = src + (size_t)slice * n;
    float*       d = dst + (size_t)slice * n;
    const int t = threadIdx.x;

    float acc = 0.f;
    for (int i = t; i < n; i += 256) acc += fabsf(s[i]);
    red[t] = acc; __syncthreads();
    for (int o = 128; o > 0; o >>= 1) { if (t < o) red[t] += red[t + o]; __syncthreads(); }
    const float delta = 0.75f * red[0] / (float)n;
    __syncthreads();

    float sa = 0.f, sc = 0.f;
    for (int i = t; i < n; i += 256) {
        float a = fabsf(s[i]);
        if (a > delta) { sa += a; sc += 1.f; }
    }
    red[t] = sa; __syncthreads();
    for (int o = 128; o > 0; o >>= 1) { if (t < o) red[t] += red[t + o]; __syncthreads(); }
    if (t == 0) stats[0] = red[0];
    __syncthreads();
    red[t] = sc; __syncthreads();
    for (int o = 128; o > 0; o >>= 1) { if (t < o) red[t] += red[t + o]; __syncthreads(); }
    if (t == 0) stats[1] = red[0];
    __syncthreads();

    const float alpha = stats[0] / fmaxf(stats[1], 1.f);
    for (int i = t; i < n; i += 256) {
        float w = s[i], a = fabsf(w);
        float sg = (w > 0.f) ? 1.f : ((w < 0.f) ? -1.f : 0.f);
        d[i] = (a > delta) ? alpha * sg : 0.f;
    }
}

// ---------------------------------------------------------------------------
// Shallow conv 3x3: 3 -> 64 channels, + bias + relu.  x: [B,3,H,W]
// ---------------------------------------------------------------------------
__global__ void shallow_conv_kernel(const float* __restrict__ x,
                                    const float* __restrict__ wq,
                                    const float* __restrict__ bias,
                                    float* __restrict__ dst) {
    size_t idx = (size_t)blockIdx.x * 256 + threadIdx.x;
    if (idx >= (size_t)BB * CC * HWHW) return;
    int w = idx & (WW - 1);
    int h = (idx >> 8) & (HH - 1);
    int co = (idx >> 16) & (CC - 1);
    int b = idx >> 22;
    float sum = bias[co];
    for (int ci = 0; ci < 3; ++ci) {
        const float* xp = x + ((size_t)(b * 3 + ci)) * HWHW;
        const float* wp = wq + (co * 3 + ci) * 9;
        for (int kh = 0; kh < 3; ++kh) {
            int ih = h + kh - 1;
            if (ih < 0 || ih >= HH) continue;
            for (int kw = 0; kw < 3; ++kw) {
                int iw = w + kw - 1;
                if (iw < 0 || iw >= WW) continue;
                sum += xp[ih * WW + iw] * wp[kh * 3 + kw];
            }
        }
    }
    dst[idx] = fmaxf(sum, 0.f);
}

// ---------------------------------------------------------------------------
// Depthwise 3x3 over 64 channels, + bias, optional relu (fp32 out, NCHW).
// ---------------------------------------------------------------------------
__global__ void dwconv3_kernel(const float* __restrict__ src,
                               const float* __restrict__ wq,
                               const float* __restrict__ bias,
                               float* __restrict__ dst, int do_relu) {
    size_t idx = (size_t)blockIdx.x * 256 + threadIdx.x;
    if (idx >= (size_t)BB * CC * HWHW) return;
    int w = idx & (WW - 1);
    int h = (idx >> 8) & (HH - 1);
    int c = (idx >> 16) & (CC - 1);
    const float* sp = src + (idx - (size_t)(h * WW + w));   // (b,c) plane start
    const float* wp = wq + c * 9;
    float sum = bias[c];
    for (int kh = 0; kh < 3; ++kh) {
        int ih = h + kh - 1;
        if (ih < 0 || ih >= HH) continue;
        for (int kw = 0; kw < 3; ++kw) {
            int iw = w + kw - 1;
            if (iw < 0 || iw >= WW) continue;
            sum += sp[ih * WW + iw] * wp[kh * 3 + kw];
        }
    }
    if (do_relu) sum = fmaxf(sum, 0.f);
    dst[idx] = sum;
}

// ---------------------------------------------------------------------------
// Depthwise 3x3 + relu, f16 output in pixel-major tiles: dst[b][h*W+w][c].
// Feeds the WMMA pointwise GEMM (halves its input stream; rows contiguous).
// ---------------------------------------------------------------------------
__global__ void dwconv3_f16t_kernel(const float* __restrict__ src,
                                    const float* __restrict__ wq,
                                    const float* __restrict__ bias,
                                    _Float16* __restrict__ dst) {
    size_t idx = (size_t)blockIdx.x * 256 + threadIdx.x;
    if (idx >= (size_t)BB * CC * HWHW) return;
    int w = idx & (WW - 1);
    int h = (idx >> 8) & (HH - 1);
    int c = (idx >> 16) & (CC - 1);
    int b = idx >> 22;
    const float* sp = src + (idx - (size_t)(h * WW + w));
    const float* wp = wq + c * 9;
    float sum = bias[c];
    for (int kh = 0; kh < 3; ++kh) {
        int ih = h + kh - 1;
        if (ih < 0 || ih >= HH) continue;
        for (int kw = 0; kw < 3; ++kw) {
            int iw = w + kw - 1;
            if (iw < 0 || iw >= WW) continue;
            sum += sp[ih * WW + iw] * wp[kh * 3 + kw];
        }
    }
    sum = fmaxf(sum, 0.f);
    dst[((size_t)b * HWHW + (size_t)(h * WW + w)) * CC + c] = (_Float16)sum;
}

// ---------------------------------------------------------------------------
// Pointwise 64 -> 8 + sigmoid (texture state).  dst: [B,8,H,W]
// ---------------------------------------------------------------------------
__global__ void pw_small_sigmoid_kernel(const float* __restrict__ src,
                                        const float* __restrict__ wq,
                                        const float* __restrict__ bias,
                                        float* __restrict__ dst) {
    size_t idx = (size_t)blockIdx.x * 256 + threadIdx.x;
    if (idx >= (size_t)BB * 8 * HWHW) return;
    int p  = idx & (HWHW - 1);
    int co = (idx >> 16) & 7;
    int b  = idx >> 19;
    const float* sp = src + (size_t)b * CC * HWHW + p;
    const float* wp = wq + co * CC;
    float sum = bias[co];
    for (int ci = 0; ci < CC; ++ci) sum += sp[(size_t)ci * HWHW] * wp[ci];
    dst[idx] = sigmoidf_(sum);
}

// ---------------------------------------------------------------------------
// Router 1x1 conv (8 -> 4) + argmax (softmax is monotone).  top: [B,H,W] int
// ---------------------------------------------------------------------------
__global__ void router_argmax_kernel(const float* __restrict__ ts,
                                     const float* __restrict__ rw,
                                     const float* __restrict__ rb,
                                     int* __restrict__ top) {
    size_t idx = (size_t)blockIdx.x * 256 + threadIdx.x;
    if (idx >= (size_t)BB * HWHW) return;
    int p = idx & (HWHW - 1);
    int b = idx >> 16;
    float s[8];
    for (int ci = 0; ci < 8; ++ci) s[ci] = ts[((size_t)(b * 8 + ci)) * HWHW + p];
    int best = 0; float bestv = -3.4e38f;
    for (int e = 0; e < NEXP; ++e) {
        float v = rb[e];
        for (int ci = 0; ci < 8; ++ci) v += rw[e * 8 + ci] * s[ci];
        if (v > bestv) { bestv = v; best = e; }
    }
    top[idx] = best;
}

// ---------------------------------------------------------------------------
// Pointwise 64x64 conv as GEMM via WMMA f32_16x16x32_f16.
// src: f16 pixel-major tiles [B, H*W, 64] (from dwconv3_f16t_kernel).
// One wave = 16-pixel strip x 64 out channels (4 M-tiles x 2 K-tiles).
// X tile (16 px * 64 ch * 2B = 2KB) staged via async copy to LDS
// (global_load_async_to_lds_b128, ASYNCcnt); weights staged with VALU
// fp32->f16 conversion.  Output written fp32 NCHW (+bias).
// ---------------------------------------------------------------------------
__global__ void pw_wmma64_kernel(const _Float16* __restrict__ src,
                                 const float* __restrict__ wq,
                                 const float* __restrict__ bias,
                                 float* __restrict__ dst) {
    __shared__ __attribute__((aligned(16))) _Float16 Wsh[64][72];    // [co][ci]
    __shared__ __attribute__((aligned(16))) _Float16 Xsh[8][16][72]; // per-wave [px][ci]
    const int tid  = threadIdx.x;
    const int wv   = tid >> 5;
    const int lane = tid & 31;

    // This wave's 16-pixel strip (strips never cross a row: W=256 % 16 == 0).
    const long long gp = (long long)blockIdx.x * 128 + (long long)wv * 16;
    const int b  = (int)(gp >> 16);
    const int p  = (int)(gp & 65535);
    const int h  = p >> 8;
    const int w0 = p & 255;

    // --- Async stage X tile: 128 x B128 transfers (4 per lane). -----------
    // Pixel n's 64 f16 channels are 128 contiguous bytes in global; LDS row
    // stride is 144B (16B-aligned, bank-conflict-free for fragment reads).
    {
        const char* gbase = (const char*)(src + ((long long)b * HWHW + p) * CC);
        const unsigned ldsbase =
            (unsigned)(uintptr_t)(&Xsh[wv][0][0]);   // low 32 bits = LDS offset
#pragma unroll
        for (int i = 0; i < 4; ++i) {
            const int t = i * 32 + lane;             // 0..127
            const int n = t >> 3, chunk = t & 7;
            const unsigned laddr = ldsbase + (unsigned)(n * 144 + chunk * 16);
            const char* gaddr = gbase + t * 16;
            asm volatile("global_load_async_to_lds_b128 %0, %1, off"
                         :: "v"(laddr), "v"(gaddr)
                         : "memory");
        }
    }

    // --- Stage ternarized weights (fp32 -> f16), shared by all 8 waves. ---
    for (int i = tid; i < 64 * 64; i += 256)
        Wsh[i >> 6][i & 63] = (_Float16)wq[i];

    asm volatile("s_wait_asynccnt 0" ::: "memory");  // this wave's X tile done
    __syncthreads();                                 // weights visible to all

    const int n    = lane & 15;
    const int half = lane >> 4;
    float* dp = dst + ((long long)b * CC) * HWHW + h * WW + w0;

    for (int mt = 0; mt < 4; ++mt) {
        const int m = mt * 16 + n;      // A-matrix row (both lane halves)
        v8f acc = {};
        for (int kt = 0; kt < 2; ++kt) {
            v16h a, bf;
#pragma unroll
            for (int j = 0; j < 8; ++j) {
                // 16-bit A/B fragment K packing (ISA 7.12.2):
                // VGPR j holds the K pair at kbase, kbase+1.
                const int kb = kt * 32 + ((j >> 2) << 4) + (half << 3) + ((j & 3) << 1);
                a[2 * j]     = Wsh[m][kb];
                a[2 * j + 1] = Wsh[m][kb + 1];
                bf[2 * j]     = Xsh[wv][n][kb];
                bf[2 * j + 1] = Xsh[wv][n][kb + 1];
            }
            acc = __builtin_amdgcn_wmma_f32_16x16x32_f16(
                false, a, false, bf, (short)0, acc, false, false);
        }
        // C/D layout: lanes 0-15 -> M=r, lanes 16-31 -> M=8+r; N = lane&15.
#pragma unroll
        for (int r = 0; r < 8; ++r) {
            const int mo = mt * 16 + half * 8 + r;
            dp[(long long)mo * HWHW + n] = acc[r] + bias[mo];
        }
    }
}

// ---------------------------------------------------------------------------
// Global average pool: one block per (b,c) plane.  y: [B*C]
// ---------------------------------------------------------------------------
__global__ void gap_kernel(const float* __restrict__ src, float* __restrict__ y) {
    __shared__ float red[256];
    const int bc = blockIdx.x;
    const float* sp = src + (size_t)bc * HWHW;
    const int t = threadIdx.x;
    float acc = 0.f;
    for (int i = t; i < HWHW; i += 256) acc += sp[i];
    red[t] = acc; __syncthreads();
    for (int o = 128; o > 0; o >>= 1) { if (t < o) red[t] += red[t + o]; __syncthreads(); }
    if (t == 0) y[bc] = red[0] * (1.0f / (float)HWHW);
}

// ---------------------------------------------------------------------------
// Channel attention FCs: y[64] -> relu(fc1) [16] -> sigmoid(fc2) [64].
// One block (64 threads) per batch element.
// ---------------------------------------------------------------------------
__global__ void attn_fc_kernel(const float* __restrict__ y,
                               const float* __restrict__ fc1,
                               const float* __restrict__ fc2,
                               float* __restrict__ scale) {
    __shared__ float y1[16];
    const int b = blockIdx.x;
    const int t = threadIdx.x;
    if (t < 16) {
        float s = 0.f;
        for (int c = 0; c < CC; ++c) s += y[b * CC + c] * fc1[t * CC + c];
        y1[t] = fmaxf(s, 0.f);
    }
    __syncthreads();
    float s = 0.f;
    for (int k = 0; k < 16; ++k) s += y1[k] * fc2[t * 16 + k];
    scale[b * CC + t] = sigmoidf_(s);
}

// ---------------------------------------------------------------------------
// out = t2 * scale[b,c] + resid   (res-block tail)
// ---------------------------------------------------------------------------
__global__ void scale_res_kernel(const float* __restrict__ t2,
                                 const float* __restrict__ scale,
                                 const float* __restrict__ resid,
                                 float* __restrict__ dst) {
    size_t idx = (size_t)blockIdx.x * 256 + threadIdx.x;
    if (idx >= (size_t)BB * CC * HWHW) return;
    int c = (idx >> 16) & (CC - 1);
    int b = idx >> 22;
    dst[idx] = t2[idx] * scale[b * CC + c] + resid[idx];
}

// ---------------------------------------------------------------------------
// Top-1 MoE select / accumulate: macc (+)= (top==e) ? cur : 0
// ---------------------------------------------------------------------------
__global__ void moe_select_kernel(const float* __restrict__ cur,
                                  const int* __restrict__ top,
                                  float* __restrict__ macc, int e, int first) {
    size_t idx = (size_t)blockIdx.x * 256 + threadIdx.x;
    if (idx >= (size_t)BB * CC * HWHW) return;
    int p = idx & (HWHW - 1);
    int b = idx >> 22;
    float v = (top[(size_t)b * HWHW + p] == e) ? cur[idx] : 0.f;
    macc[idx] = first ? v : (macc[idx] + v);
}

// ---------------------------------------------------------------------------
// Reconstruction conv 3x3: 64 -> 12 (plain fp32 weights), + bias.
// ---------------------------------------------------------------------------
__global__ void recon_conv_kernel(const float* __restrict__ src,
                                  const float* __restrict__ wgt,
                                  const float* __restrict__ bias,
                                  float* __restrict__ dst) {
    size_t idx = (size_t)blockIdx.x * 256 + threadIdx.x;
    if (idx >= (size_t)BB * 12 * HWHW) return;
    int w  = idx & (WW - 1);
    int h  = (idx >> 8) & (HH - 1);
    int co = (int)((idx >> 16) % 12);
    int b  = (int)(idx / ((size_t)12 * HWHW));
    float sum = bias[co];
    for (int ci = 0; ci < CC; ++ci) {
        const float* sp = src + ((size_t)(b * CC + ci)) * HWHW;
        const float* wp = wgt + (co * CC + ci) * 9;
        for (int kh = 0; kh < 3; ++kh) {
            int ih = h + kh - 1;
            if (ih < 0 || ih >= HH) continue;
            for (int kw = 0; kw < 3; ++kw) {
                int iw = w + kw - 1;
                if (iw < 0 || iw >= WW) continue;
                sum += sp[ih * WW + iw] * wp[kh * 3 + kw];
            }
        }
    }
    dst[idx] = sum;
}

// ---------------------------------------------------------------------------
// Pixel shuffle (r=2) + half-pixel bilinear 2x upsample of x, summed.
// recon: [B,12,H,W]; x: [B,3,H,W]; sr: [B,3,2H,2W]
// ---------------------------------------------------------------------------
__global__ void ps_upsample_kernel(const float* __restrict__ recon,
                                   const float* __restrict__ x,
                                   float* __restrict__ sr) {
    size_t idx = (size_t)blockIdx.x * 256 + threadIdx.x;
    if (idx >= (size_t)BB * 3 * H2 * W2) return;
    int ow = idx & (W2 - 1);
    int oh = (int)((idx >> 9) & (H2 - 1));
    int c  = (int)((idx / ((size_t)H2 * W2)) % 3);
    int b  = (int)(idx / ((size_t)3 * H2 * W2));

    int rc = c * 4 + (oh & 1) * 2 + (ow & 1);
    float v = recon[((size_t)(b * 12 + rc)) * HWHW + (oh >> 1) * WW + (ow >> 1)];

    float sy = (oh + 0.5f) * 0.5f - 0.5f;
    float sx = (ow + 0.5f) * 0.5f - 0.5f;
    sy = fminf(fmaxf(sy, 0.f), (float)(HH - 1));
    sx = fminf(fmaxf(sx, 0.f), (float)(WW - 1));
    int y0 = (int)sy, x0 = (int)sx;
    int y1 = min(y0 + 1, HH - 1), x1 = min(x0 + 1, WW - 1);
    float fy = sy - (float)y0, fx = sx - (float)x0;
    const float* xp = x + ((size_t)(b * 3 + c)) * HWHW;
    float bl = (1.f - fy) * ((1.f - fx) * xp[y0 * WW + x0] + fx * xp[y0 * WW + x1]) +
               fy         * ((1.f - fx) * xp[y1 * WW + x0] + fx * xp[y1 * WW + x1]);
    sr[idx] = v + bl;
}

// ---------------------------------------------------------------------------
// 3-channel 3x3 conv on the 512x512 SR image (ternarized refinement convs).
// Optional relu; optional elementwise add of `addsrc` (residual skip).
// ---------------------------------------------------------------------------
__global__ void conv3x3_c3_kernel(const float* __restrict__ src,
                                  const float* __restrict__ wq,
                                  const float* __restrict__ bias,
                                  const float* __restrict__ addsrc,
                                  float* __restrict__ dst, int do_relu) {
    size_t idx = (size_t)blockIdx.x * 256 + threadIdx.x;
    if (idx >= (size_t)BB * 3 * H2 * W2) return;
    int w  = idx & (W2 - 1);
    int h  = (int)((idx >> 9) & (H2 - 1));
    int co = (int)((idx / ((size_t)H2 * W2)) % 3);
    int b  = (int)(idx / ((size_t)3 * H2 * W2));
    float sum = bias[co];
    for (int ci = 0; ci < 3; ++ci) {
        const float* sp = src + ((size_t)(b * 3 + ci)) * H2 * W2;
        const float* wp = wq + (co * 3 + ci) * 9;
        for (int kh = 0; kh < 3; ++kh) {
            int ih = h + kh - 1;
            if (ih < 0 || ih >= H2) continue;
            for (int kw = 0; kw < 3; ++kw) {
                int iw = w + kw - 1;
                if (iw < 0 || iw >= W2) continue;
                sum += sp[(size_t)ih * W2 + iw] * wp[kh * 3 + kw];
            }
        }
    }
    if (do_relu) sum = fmaxf(sum, 0.f);
    if (addsrc)  sum += addsrc[idx];
    dst[idx] = sum;
}

// ---------------------------------------------------------------------------
extern "C" void kernel_launch(void* const* d_in, const int* in_sizes, int n_in,
                              void* d_out, int out_size, void* d_ws, size_t ws_size,
                              hipStream_t stream) {
    const float* x         = (const float*)d_in[0];
    const float* shallow_w = (const float*)d_in[1];
    const float* shallow_b = (const float*)d_in[2];
    const float* tex_dw_w  = (const float*)d_in[3];
    const float* tex_dw_b  = (const float*)d_in[4];
    const float* tex_pw_w  = (const float*)d_in[5];
    const float* tex_pw_b  = (const float*)d_in[6];
    const float* router_w  = (const float*)d_in[7];
    const float* router_b  = (const float*)d_in[8];
    const float* exp_dw_w  = (const float*)d_in[9];
    const float* exp_dw_b  = (const float*)d_in[10];
    const float* exp_pw_w  = (const float*)d_in[11];
    const float* exp_pw_b  = (const float*)d_in[12];
    const float* exp_fc1_w = (const float*)d_in[13];
    const float* exp_fc2_w = (const float*)d_in[14];
    const float* recon_w   = (const float*)d_in[15];
    const float* recon_b   = (const float*)d_in[16];
    const float* ref1_w    = (const float*)d_in[17];
    const float* ref1_b    = (const float*)d_in[18];
    const float* ref2_w    = (const float*)d_in[19];
    const float* ref2_b    = (const float*)d_in[20];
    float* out = (float*)d_out;

    const size_t NF = (size_t)BB * CC * HWHW;       // 16.78M floats (64MB)
    const size_t NT = (size_t)BB * 8 * HWHW;        // texture state
    const size_t NP = (size_t)BB * HWHW;            // pixels
    const size_t NS = (size_t)BB * 3 * H2 * W2;     // SR image

    float* ws = (float*)d_ws;
    size_t off = 0;
    // 64-float (256B) aligned carve-outs so f16 tile rows stay 16B-aligned.
    auto take = [&](size_t n) {
        float* p = ws + off;
        off += (n + 63) & ~(size_t)63;
        return p;
    };

    float* qw_shallow = take(64 * 27);
    float* qw_texdw   = take(64 * 9);
    float* qw_texpw   = take(8 * 64);
    float* qw_expdw   = take(NEXP * NBLK * 64 * 9);
    float* qw_exppw   = take(NEXP * NBLK * 64 * 64);
    float* qw_ref1    = take(81);
    float* qw_ref2    = take(81);
    float* feat       = take(NF);
    float* buf1       = take(NF);           // texture t / recon output
    float* buf2       = take(NF);           // pointwise outputs
    float* cur        = take(NF);           // expert running activation
    float* macc       = take(NF);           // MoE-selected output
    _Float16* tbuf    = (_Float16*)take(NF / 2);  // dw f16 pixel-major tiles
    float* tstate     = take(NT);
    int*   top        = (int*)take(NP);
    float* yv         = take(BB * CC);
    float* scale      = take(BB * CC);
    float* sr         = take(NS);
    float* rbuf       = take(NS);
    (void)ws_size; (void)in_sizes; (void)n_in; (void)out_size;

    const int TB = 256;
    const int GF = (int)(NF / TB);          // 65536 blocks
    const int GT = (int)(NT / TB);
    const int GP = (int)(NP / TB);
    const int GS = (int)(NS / TB);

    // 1) Ternarize all _fpsan_conv weights (per reference slice granularity).
    ternarize_kernel<<<1, 256, 0, stream>>>(shallow_w, qw_shallow, 64 * 27);
    ternarize_kernel<<<1, 256, 0, stream>>>(tex_dw_w, qw_texdw, 64 * 9);
    ternarize_kernel<<<1, 256, 0, stream>>>(tex_pw_w, qw_texpw, 8 * 64);
    ternarize_kernel<<<NEXP * NBLK, 256, 0, stream>>>(exp_dw_w, qw_expdw, 64 * 9);
    ternarize_kernel<<<NEXP * NBLK, 256, 0, stream>>>(exp_pw_w, qw_exppw, 64 * 64);
    ternarize_kernel<<<1, 256, 0, stream>>>(ref1_w, qw_ref1, 81);
    ternarize_kernel<<<1, 256, 0, stream>>>(ref2_w, qw_ref2, 81);

    // 2) Shallow features.
    shallow_conv_kernel<<<GF, TB, 0, stream>>>(x, qw_shallow, shallow_b, feat);

    // 3) Texture state + router top-1.
    dwconv3_kernel<<<GF, TB, 0, stream>>>(feat, qw_texdw, tex_dw_b, buf1, 1);
    pw_small_sigmoid_kernel<<<GT, TB, 0, stream>>>(buf1, qw_texpw, tex_pw_b, tstate);
    router_argmax_kernel<<<GP, TB, 0, stream>>>(tstate, router_w, router_b, top);

    // 4) Experts (dense, as in eval-mode reference), WMMA pointwise GEMMs
    //    fed by f16 pixel-major tiles staged to LDS with async copies.
    for (int e = 0; e < NEXP; ++e) {
        for (int nb = 0; nb < NBLK; ++nb) {
            const int sl = e * NBLK + nb;
            const float* inbuf = (nb == 0) ? feat : cur;
            dwconv3_f16t_kernel<<<GF, TB, 0, stream>>>(
                inbuf, qw_expdw + (size_t)sl * 64 * 9, exp_dw_b + sl * 64, tbuf);
            pw_wmma64_kernel<<<(int)(NP / 128), TB, 0, stream>>>(
                tbuf, qw_exppw + (size_t)sl * 64 * 64, exp_pw_b + sl * 64, buf2);
            gap_kernel<<<BB * CC, 256, 0, stream>>>(buf2, yv);
            attn_fc_kernel<<<BB, 64, 0, stream>>>(
                yv, exp_fc1_w + (size_t)sl * 16 * 64, exp_fc2_w + (size_t)sl * 64 * 16, scale);
            scale_res_kernel<<<GF, TB, 0, stream>>>(buf2, scale, inbuf, cur);
        }
        moe_select_kernel<<<GF, TB, 0, stream>>>(cur, top, macc, e, e == 0);
    }

    // 5) Reconstruction + pixel shuffle + bilinear skip.
    recon_conv_kernel<<<(int)((size_t)BB * 12 * HWHW / TB), TB, 0, stream>>>(
        macc, recon_w, recon_b, buf1);
    ps_upsample_kernel<<<GS, TB, 0, stream>>>(buf1, x, sr);

    // 6) Ternary refinement: out = sr + ref2(relu(ref1(sr)))
    conv3x3_c3_kernel<<<GS, TB, 0, stream>>>(sr, qw_ref1, ref1_b, nullptr, rbuf, 1);
    conv3x3_c3_kernel<<<GS, TB, 0, stream>>>(rbuf, qw_ref2, ref2_b, sr, out, 0);
}
